// Attention_23021024706744
// MI455X (gfx1250) — matrix operand order repaired and checked by reading, not compiled, not run on hardware
//
#include <hip/hip_runtime.h>
#include <hip/hip_bf16.h>
#include <math.h>

typedef __attribute__((ext_vector_type(16))) _Float16 v16h;
typedef __attribute__((ext_vector_type(8)))  float    v8f;

#define ATTN_B  2
#define ATTN_L  2048
#define ATTN_D  1024
#define ATTN_H  16
#define ATTN_HD 64
#define NEG_BIG (-3.0e38f)

__device__ __forceinline__ v8f wmma_f16(v16h a, v16h b, v8f c) {
  // D = A(16x32 f16) * B(32x16 f16) + C(16x16 f32)
  return __builtin_amdgcn_wmma_f32_16x16x32_f16(
      /*neg_a=*/false, a, /*neg_b=*/false, b,
      /*c_mod=*/(short)0, c, /*reuse_a=*/false, /*reuse_b=*/false);
}

// ---- fragment loaders (f16 source, all contiguous 16B/32B per lane) --------
// A-tile 16x32 (M x K): lane r (0..15) row m0+r; elems 0..7 = K(0..7)+h*8,
// elems 8..15 = K(16..23)+h*8  (h = lane>=16)
__device__ __forceinline__ v16h load_a_f16(const _Float16* __restrict__ X, int ld,
                                           int m0, int k0, int lane) {
  int r = lane & 15, h = (lane >> 4) & 1;
  const _Float16* p = X + (size_t)(m0 + r) * ld + k0 + h * 8;
  v16h a;
#pragma unroll
  for (int i = 0; i < 8; ++i) { a[i] = p[i]; a[8 + i] = p[16 + i]; }
  return a;
}

// B-tile 32x16 (K x N) where B[k][n] = S[n0+n][k0+k], S row-major [rows, ld].
// Lane n holds column n: elems 0..7 = K(0..7)+h*8, elems 8..15 = K(16..23)+h*8 (h = lane>=16).
__device__ __forceinline__ v16h load_bT_f16(const _Float16* __restrict__ S, int ld,
                                            int n0, int k0, int lane) {
  int n = lane & 15, h = (lane >> 4) & 1;
  const _Float16* p = S + (size_t)(n0 + n) * ld + k0 + h * 8;
  v16h b;
#pragma unroll
  for (int i = 0; i < 8; ++i) { b[i] = p[i]; b[8 + i] = p[16 + i]; }
  return b;
}

// ---- elementwise fp32 -> f16 (one pass; removes cvt work from GEMM loops) --
__global__ __launch_bounds__(256) void cvt_f16_kernel(const float* __restrict__ in,
                                                      _Float16* __restrict__ out,
                                                      int n8) {
  int i = blockIdx.x * blockDim.x + threadIdx.x;
  if (i >= n8) return;
  size_t base = (size_t)i * 8;
#pragma unroll
  for (int j = 0; j < 8; ++j) out[base + j] = (_Float16)in[base + j];
}

// ---- generic NT GEMM: out = A @ W^T + bias, all-f16 operands ---------------
// Wave tile 32x64: 8 WMMAs per 32-wide k-step, B fragments shared across 2 M rows.
// mode 0: f16 out, head-split [B,H,L,HD]     (Q, K projections)
// mode 1: f16 out, transposed [B,H,HD,L]     (V projection; makes P@V B-frags contiguous)
// mode 2: f32 out, row-major  [M, D]         (final output projection)
__global__ __launch_bounds__(256) void gemm_nt_f16(
    const _Float16* __restrict__ A,   // [M, K] f16
    const _Float16* __restrict__ W,   // [N=D, K] f16
    const float* __restrict__ bias,   // [D] fp32
    void* __restrict__ outv,
    int K, int mode) {
  int lane = threadIdx.x & 31;
  int wave = threadIdx.x >> 5;
  int m0 = (blockIdx.y * 8 + wave) * 32;
  int n0 = blockIdx.x * 64;
  int cl = lane & 15;

  const v8f vzero = {0.f, 0.f, 0.f, 0.f, 0.f, 0.f, 0.f, 0.f};
  v8f acc[2][4];
#pragma unroll
  for (int s = 0; s < 2; ++s)
#pragma unroll
    for (int t = 0; t < 4; ++t) acc[s][t] = vzero;

  for (int k0 = 0; k0 < K; k0 += 32) {
    if (k0 + 64 < K) {  // prefetch next-next B k-tile (global_prefetch_b8)
      __builtin_prefetch((const void*)(W + (size_t)(n0 + cl) * K + k0 + 64), 0, 3);
    }
    v16h a0 = load_a_f16(A, K, m0, k0, lane);
    v16h a1 = load_a_f16(A, K, m0 + 16, k0, lane);
#pragma unroll
    for (int t = 0; t < 4; ++t) {
      v16h b = load_bT_f16(W, K, n0 + 16 * t, k0, lane);
      acc[0][t] = wmma_f16(a0, b, acc[0][t]);
      acc[1][t] = wmma_f16(a1, b, acc[1][t]);
    }
  }

  int hh = (lane >> 4) & 1;
#pragma unroll
  for (int s = 0; s < 2; ++s) {
    int mbase = m0 + 16 * s;
#pragma unroll
    for (int t = 0; t < 4; ++t) {
      int n = n0 + 16 * t + cl;
      float bn = bias[n];
      int hidx = n >> 6, hd = n & (ATTN_HD - 1);
#pragma unroll
      for (int i = 0; i < 8; ++i) {
        int m = mbase + i + 8 * hh;        // m = b*L + l
        float v = acc[s][t][i] + bn;
        if (mode == 2) {
          ((float*)outv)[(size_t)m * ATTN_D + n] = v;
        } else {
          int bidx = m >> 11, l = m & (ATTN_L - 1);
          size_t idx;
          if (mode == 0)
            idx = (((size_t)bidx * ATTN_H + hidx) * ATTN_L + l) * ATTN_HD + hd;
          else
            idx = (((size_t)bidx * ATTN_H + hidx) * ATTN_HD + hd) * ATTN_L + l;
          ((_Float16*)outv)[idx] = (_Float16)v;
        }
      }
    }
  }
}

// ---- Flash attention: one wave owns a 16-row Q tile, streams causal K/V ----
__global__ __launch_bounds__(256) void attn_kernel(
    const _Float16* __restrict__ qh,   // [B,H,L,HD]
    const _Float16* __restrict__ kh,   // [B,H,L,HD]
    const _Float16* __restrict__ vt,   // [B,H,HD,L]
    const unsigned char* __restrict__ kmask, // [B,L] bool
    _Float16* __restrict__ out,        // [B,L,D]
    float scale) {
  __shared__ _Float16 plds[8][16][32]; // per-wave P staging (C-layout -> A-layout)

  int lane = threadIdx.x & 31;
  int wave = threadIdx.x >> 5;
  int b = blockIdx.z, h = blockIdx.y;
  int qbase = blockIdx.x * 128 + wave * 16;
  int hh = (lane >> 4) & 1;
  int cl = lane & 15;

  const _Float16* qb = qh + (((size_t)b * ATTN_H + h) * ATTN_L) * ATTN_HD;
  const _Float16* kb = kh + (((size_t)b * ATTN_H + h) * ATTN_L) * ATTN_HD;
  const _Float16* vb = vt + (((size_t)b * ATTN_H + h) * ATTN_HD) * ATTN_L;
  const unsigned char* kmb = kmask + (size_t)b * ATTN_L;

  // Q A-fragments, loaded once (HD=64 -> two K-chunks of 32)
  v16h qa0 = load_a_f16(qb, ATTN_HD, qbase, 0, lane);
  v16h qa1 = load_a_f16(qb, ATTN_HD, qbase, 32, lane);

  const v8f vzero = {0.f, 0.f, 0.f, 0.f, 0.f, 0.f, 0.f, 0.f};
  v8f accv[4];
#pragma unroll
  for (int t = 0; t < 4; ++t) accv[t] = vzero;
  float mrow[8], lrow[8];
#pragma unroll
  for (int i = 0; i < 8; ++i) { mrow[i] = NEG_BIG; lrow[i] = 0.f; }

  int kend = qbase + 16;                 // causal: only keys <= last row of tile
  for (int kc = 0; kc < kend; kc += 32) {
    // S tile (16 rows x 32 keys): two 16x16 WMMA accumulators, 2 K-steps each
    v8f s0 = vzero, s1 = vzero;
    {
      v16h b0 = load_bT_f16(kb, ATTN_HD, kc,      0,  lane);
      s0 = wmma_f16(qa0, b0, s0);
      v16h b1 = load_bT_f16(kb, ATTN_HD, kc,      32, lane);
      s0 = wmma_f16(qa1, b1, s0);
      v16h b2 = load_bT_f16(kb, ATTN_HD, kc + 16, 0,  lane);
      s1 = wmma_f16(qa0, b2, s1);
      v16h b3 = load_bT_f16(kb, ATTN_HD, kc + 16, 32, lane);
      s1 = wmma_f16(qa1, b3, s1);
    }

    int c0 = kc + cl;
    int c1 = kc + 16 + cl;
    bool km0 = kmb[c0] != 0;
    bool km1 = kmb[c1] != 0;

    float p0[8], p1[8], rsc[8];
#pragma unroll
    for (int i = 0; i < 8; ++i) {
      int row = qbase + i + 8 * hh;
      float v0 = (km0 && c0 <= row) ? s0[i] * scale : NEG_BIG;
      float v1 = (km1 && c1 <= row) ? s1[i] * scale : NEG_BIG;
      float cm = fmaxf(v0, v1);
      // row-max across the 16 lanes (columns) of this half-wave
      cm = fmaxf(cm, __shfl_xor(cm, 1, 16));
      cm = fmaxf(cm, __shfl_xor(cm, 2, 16));
      cm = fmaxf(cm, __shfl_xor(cm, 4, 16));
      cm = fmaxf(cm, __shfl_xor(cm, 8, 16));
      float mo = mrow[i];
      float mn = fmaxf(mo, cm);
      float rr = (mo > -1.0e37f) ? __expf(mo - mn) : 0.0f;
      float e0 = (v0 > -1.0e37f) ? __expf(v0 - mn) : 0.0f;
      float e1 = (v1 > -1.0e37f) ? __expf(v1 - mn) : 0.0f;
      mrow[i] = mn;
      lrow[i] = lrow[i] * rr + e0 + e1;   // per-lane partial; reduced at the end
      rsc[i] = rr;
      p0[i] = e0; p1[i] = e1;
    }
#pragma unroll
    for (int t = 0; t < 4; ++t)
#pragma unroll
      for (int i = 0; i < 8; ++i) accv[t][i] *= rsc[i];

    // stage P (f16) through this wave's private LDS to convert C-layout -> A-layout
#pragma unroll
    for (int i = 0; i < 8; ++i) {
      int r = i + 8 * hh;
      plds[wave][r][cl]      = (_Float16)p0[i];
      plds[wave][r][16 + cl] = (_Float16)p1[i];
    }
    asm volatile("s_wait_dscnt 0" ::: "memory");  // same-wave LDS ops are in-order
    v16h pa;
    {
      const _Float16* pr = &plds[wave][cl][hh * 8];
#pragma unroll
      for (int i = 0; i < 8; ++i) { pa[i] = pr[i]; pa[8 + i] = pr[16 + i]; }
    }

    // O += P(16x32) @ V(32x64): 4 d-tiles, V^T layout makes B-fragments contiguous
#pragma unroll
    for (int t = 0; t < 4; ++t) {
      v16h vbt = load_bT_f16(vb, ATTN_L, 16 * t, kc, lane);
      accv[t] = wmma_f16(pa, vbt, accv[t]);
    }
  }

  // finalize: reduce row-sums across 16 lanes, normalize, write [B,L,D] f16
#pragma unroll
  for (int i = 0; i < 8; ++i) {
    float lv = lrow[i];
    lv += __shfl_xor(lv, 1, 16);
    lv += __shfl_xor(lv, 2, 16);
    lv += __shfl_xor(lv, 4, 16);
    lv += __shfl_xor(lv, 8, 16);
    lrow[i] = (lv > 0.f) ? (1.0f / lv) : 0.0f;
  }
#pragma unroll
  for (int t = 0; t < 4; ++t)
#pragma unroll
    for (int i = 0; i < 8; ++i) {
      int lpos = qbase + i + 8 * hh;
      int d = h * ATTN_HD + 16 * t + cl;
      out[((size_t)b * ATTN_L + lpos) * ATTN_D + d] = (_Float16)(accv[t][i] * lrow[i]);
    }
}

extern "C" void kernel_launch(void* const* d_in, const int* in_sizes, int n_in,
                              void* d_out, int out_size, void* d_ws, size_t ws_size,
                              hipStream_t stream) {
  (void)in_sizes; (void)n_in; (void)out_size; (void)ws_size;
  const float* q  = (const float*)d_in[0];
  const float* k  = (const float*)d_in[1];
  const float* v  = (const float*)d_in[2];
  const unsigned char* kmask = (const unsigned char*)d_in[3];
  const float* Wq = (const float*)d_in[4];
  const float* bq = (const float*)d_in[5];
  const float* Wk = (const float*)d_in[6];
  const float* bk = (const float*)d_in[7];
  const float* Wv = (const float*)d_in[8];
  const float* bv = (const float*)d_in[9];
  const float* Wp = (const float*)d_in[10];
  const float* bp = (const float*)d_in[11];
  float* out = (float*)d_out;

  const int M = ATTN_B * ATTN_L;                       // 4096
  const size_t nx = (size_t)M * ATTN_D;                // 4M elems per activation
  const size_t nw = (size_t)ATTN_D * ATTN_D;           // 1M elems per weight

  // f16 workspace layout (56 MB total):
  _Float16* xq = (_Float16*)d_ws;     // converted activations
  _Float16* xk = xq + nx;
  _Float16* xv = xk + nx;
  _Float16* wq = xv + nx;             // converted weights
  _Float16* wk = wq + nw;
  _Float16* wv = wk + nw;
  _Float16* wp = wv + nw;
  _Float16* qh = wp + nw;             // projected per-head tensors
  _Float16* kh = qh + nx;
  _Float16* vt = kh + nx;
  _Float16* ao = xq;                  // attn output aliases xq (dead after projections)

  // 1) one-shot fp32 -> f16 conversion of activations and weights
  {
    int nx8 = (int)(nx / 8), nw8 = (int)(nw / 8);
    dim3 bx(256);
    cvt_f16_kernel<<<dim3((nx8 + 255) / 256), bx, 0, stream>>>(q, xq, nx8);
    cvt_f16_kernel<<<dim3((nx8 + 255) / 256), bx, 0, stream>>>(k, xk, nx8);
    cvt_f16_kernel<<<dim3((nx8 + 255) / 256), bx, 0, stream>>>(v, xv, nx8);
    cvt_f16_kernel<<<dim3((nw8 + 255) / 256), bx, 0, stream>>>(Wq, wq, nw8);
    cvt_f16_kernel<<<dim3((nw8 + 255) / 256), bx, 0, stream>>>(Wk, wk, nw8);
    cvt_f16_kernel<<<dim3((nw8 + 255) / 256), bx, 0, stream>>>(Wv, wv, nw8);
    cvt_f16_kernel<<<dim3((nw8 + 255) / 256), bx, 0, stream>>>(Wp, wp, nw8);
  }

  // 2) QKV projections (pure f16 WMMA GEMMs, 32x64 wave tiles)
  dim3 gproj(ATTN_D / 64, M / 256);                    // (16, 16)
  gemm_nt_f16<<<gproj, 256, 0, stream>>>(xq, wq, bq, qh, ATTN_D, 0);
  gemm_nt_f16<<<gproj, 256, 0, stream>>>(xk, wk, bk, kh, ATTN_D, 0);
  gemm_nt_f16<<<gproj, 256, 0, stream>>>(xv, wv, bv, vt, ATTN_D, 1);

  // 3) causal flash attention
  dim3 gat(ATTN_L / 128, ATTN_H, ATTN_B);              // (16, 16, 2)
  attn_kernel<<<gat, 256, 0, stream>>>(qh, kh, vt, kmask, ao, 0.125f);

  // 4) output projection -> fp32
  gemm_nt_f16<<<gproj, 256, 0, stream>>>(ao, wp, bp, out, ATTN_D, 2);
}
